// MultiheadAttention_Conv_26044681682982
// MI455X (gfx1250) — compile-verified, matrix operand order checked by reading, not run on hardware
//
#include <hip/hip_runtime.h>
#include <math.h>

#define DEV __device__ __forceinline__

typedef __attribute__((ext_vector_type(16))) __bf16 v16bf;
typedef __attribute__((ext_vector_type(8)))  float  v8f;

// ---------------- problem constants ----------------
constexpr int B_   = 2;
constexpr int N_   = 256;
constexpr int P_   = 16;
constexpr int DM_  = 256;
constexpr int H_   = 8;
constexpr int DK_  = 32;
constexpr int BN   = B_ * N_;          // 512
constexpr int S_   = P_ * P_;          // 256 spatial samples per patch
constexpr int KMIX  = 2 * DM_;         // 512: [re(c) | im(c)]
constexpr int NMIX  = H_ * 2 * DK_;    // 512: h * {re,im} * dk
constexpr int FEAT  = DK_ * S_;        // 8192 attention feature width
constexpr float INV_SCALE = 1.0f / 256.0f;     // 1/(P*P)
constexpr float PI2_16 = 0.39269908169872414f; // 2*pi/16

// ---------------- workspace layout (bytes), assumes ws_size >= ~456 MB ----------------
constexpr size_t MBy = 1ull << 20;
constexpr size_t OFF_XT    = 0;          // bf16 [131072][256]   64MB (reused as Feat later)
constexpr size_t OFF_XFT   = 64 * MBy;   // bf16 [128][512][512] 64MB
constexpr size_t OFF_WBIG  = 128 * MBy;  // bf16 [128][512][512] 64MB (per spectral, reused)
constexpr size_t OFF_YBF   = 192 * MBy;  // bf16 [131072][256]   64MB (per spectral, reused)
constexpr size_t OFF_QSP   = 256 * MBy;  // bf16 [16][256][8192] 64MB
constexpr size_t OFF_KSP   = 320 * MBy;  // bf16 [16][256][8192] 64MB
constexpr size_t OFF_VSPT  = 384 * MBy;  // bf16 [16][8192][256] 64MB (B-transposed for D2)
constexpr size_t OFF_SCORE = 448 * MBy;  // f32  [16][256][256]  4MB
constexpr size_t OFF_ATTN  = 452 * MBy;  // bf16 [16][256][256]  2MB
constexpr size_t OFF_FWDT  = 454 * MBy;  // bf16 [256][256] forward DFT basis (B^T)
constexpr size_t OFF_BAST  = OFF_FWDT + 256 * 256 * 2;  // bf16 [256][256] inverse basis (B^T)
constexpr size_t OFF_WOUT  = OFF_BAST + 256 * 256 * 2;  // bf16 [256][256] w_out ([cout][f])
constexpr size_t OFF_FEAT  = OFF_XT;     // reuse xT region

DEV v8f wmma_bf16(v16bf a, v16bf b, v8f c) {
  return __builtin_amdgcn_wmma_f32_16x16x32_bf16(false, a, false, b, (short)0, c, false, false);
}

// ---------------- register-blocked WMMA: one wave -> 32(M) x 64(N) ----------------
// A row-major [M][lda]; BT row-major [N][ldb] (B transposed -> contiguous 32B lane frags).
DEV void wmma_block2x4(const __bf16* __restrict__ A, int lda,
                       const __bf16* __restrict__ BT, int ldb,
                       int K, int lane, v8f acc[2][4]) {
  const int mn = lane & 15;
  const int kh = (lane >> 4) << 4;
  const __bf16* ap0 = A + (size_t)mn * lda + kh;
  const __bf16* ap1 = ap0 + (size_t)16 * lda;
  const __bf16* bp  = BT + (size_t)mn * ldb + kh;
  for (int k = 0; k < K; k += 32) {
    __builtin_prefetch(ap0 + k + 512, 0, 1);   // global_prefetch_b8
    v16bf a0 = *(const v16bf*)(ap0 + k);
    v16bf a1 = *(const v16bf*)(ap1 + k);
#pragma unroll
    for (int ni = 0; ni < 4; ++ni) {
      v16bf b = *(const v16bf*)(bp + (size_t)ni * 16 * ldb + k);
      acc[0][ni] = wmma_bf16(a0, b, acc[0][ni]);
      acc[1][ni] = wmma_bf16(a1, b, acc[1][ni]);
    }
  }
}

DEV void zero_acc(v8f acc[2][4]) {
  const v8f z = {0.f, 0.f, 0.f, 0.f, 0.f, 0.f, 0.f, 0.f};
#pragma unroll
  for (int mi = 0; mi < 2; ++mi)
#pragma unroll
    for (int ni = 0; ni < 4; ++ni) acc[mi][ni] = z;
}

// ---------------- init: DFT bases + w_out pack ----------------
__global__ void init_basis(const float* __restrict__ w_out,
                           __bf16* __restrict__ FwdT,
                           __bf16* __restrict__ BasT,
                           __bf16* __restrict__ WoutT) {
  const int row = blockIdx.x, col = threadIdx.x;
  {
    const int slot = row >> 1, part = row & 1;
    const int kx = slot >> 3, ky = slot & 7;
    const int px = col >> 4, py = col & 15;
    const float th = PI2_16 * (float)(kx * px + ky * py);
    FwdT[row * 256 + col] = (__bf16)(part ? -__sinf(th) : __cosf(th));
  }
  {
    const int slot = col >> 1, part = col & 1;
    const int kx = slot >> 3, ky = slot & 7;
    const int px = row >> 4, py = row & 15;
    const float th = PI2_16 * (float)(kx * px + ky * py);
    const float wgt = (ky == 0) ? 1.0f : 2.0f;
    BasT[row * 256 + col] = (__bf16)((1.0f / 256.0f) * wgt * (part ? -__sinf(th) : __cosf(th)));
  }
  WoutT[row * 256 + col] = (__bf16)w_out[row * 256 + col];
}

// ---------------- transpose x -> xT (bf16): [bn][s][c] -> [bn*DM+c][s] ----------------
__global__ void transpose_x(const float* __restrict__ x, __bf16* __restrict__ xT) {
  const int bn = blockIdx.x, c = threadIdx.x;
  const float* src = x + (size_t)bn * S_ * DM_ + c;
  __bf16* dst = xT + ((size_t)bn * DM_ + c) * S_;
  for (int s = 0; s < S_; ++s) dst[s] = (__bf16)src[(size_t)s * DM_];
}

// ---------------- pack spectral weights: complex -> real block GEMM form ----------------
// WbigT[slot][nn][kk], nn = h*64 + opart*32 + o, kk = kpart*256 + i
__global__ void pack_w(const float* __restrict__ w1, const float* __restrict__ w2,
                       __bf16* __restrict__ WbigT) {
  const size_t idx = (size_t)blockIdx.x * 256 + threadIdx.x;
  const int kk   = (int)(idx & 511);
  const int nn   = (int)((idx >> 9) & 511);
  const int slot = (int)(idx >> 18);
  const int kpart = kk >> 8, i = kk & 255;
  const int h = nn >> 6, opart = (nn >> 5) & 1, o = nn & 31;
  const int m1 = (slot >> 3) & 7, m2 = slot & 7;
  const float* w = (slot < 64) ? w1 : w2;
  const size_t base = ((((((size_t)i * DK_ + o) * 8 + m1) * 8 + m2) * H_ + h) << 1);
  const float wre = w[base], wim = w[base + 1];
  float v;
  if (opart == 0) v = (kpart == 0) ? wre : -wim;
  else            v = (kpart == 0) ? wim :  wre;
  WbigT[idx] = (__bf16)v;
}

// ---------------- stage A: forward DFT GEMM [131072x256]@[256x256] ----------------
__global__ void stageA_fft(const __bf16* __restrict__ xT,
                           const __bf16* __restrict__ FwdT,
                           __bf16* __restrict__ Xft) {
  const int w = blockIdx.x * (blockDim.x >> 5) + (threadIdx.x >> 5);
  const int lane = threadIdx.x & 31;
  const int nm = w & 3, mm = w >> 2;               // 4096 x 4 macro tiles (32x64)
  v8f acc[2][4]; zero_acc(acc);
  wmma_block2x4(xT + (size_t)mm * 32 * S_, S_, FwdT + (size_t)nm * 64 * 256, 256, 256, lane, acc);
  const int rb = (lane >> 4) << 3;
#pragma unroll
  for (int ni = 0; ni < 4; ++ni) {
    const int col = nm * 64 + ni * 16 + (lane & 15);
    const int slot = col >> 1, part = col & 1;
#pragma unroll
    for (int mi = 0; mi < 2; ++mi)
#pragma unroll
      for (int r = 0; r < 8; ++r) {
        const int row = mm * 32 + mi * 16 + rb + r;       // (bn, c)
        const int bn = row >> 8, c = row & 255;
        Xft[((size_t)slot * BN + bn) * KMIX + part * DM_ + c] = (__bf16)acc[mi][ni][r];
      }
  }
}

// ---------------- stage B: per-mode complex channel mixing, LDS-staged B panel ------
// 128 GEMMs of 512x512x512. Block = 8 waves -> 256(M) x 64(N); B panel (64 cols x 512 K)
// staged in LDS via async global->LDS copies in two K halves, read back as ds_load_b128.
constexpr int BPAD = 272;  // padded LDS row length (elements) for a 256-wide K half
__global__ void stageB_mix(const __bf16* __restrict__ Xft,
                           const __bf16* __restrict__ WbigT,
                           __bf16* __restrict__ Ybf) {
  __shared__ __align__(128) __bf16 Bpan[64 * BPAD];      // ~34KB
  const int wv = threadIdx.x >> 5, lane = threadIdx.x & 31;
  const int nb = blockIdx.x & 7, mb = (blockIdx.x >> 3) & 1, slot = blockIdx.x >> 4;
  const __bf16* bsrc = WbigT + ((size_t)slot * NMIX + nb * 64) * KMIX;
  const uint32_t lbase = (uint32_t)(uintptr_t)(&Bpan[0]);

  const int mn = lane & 15, kh = (lane >> 4) << 4;
  const __bf16* ap0 = Xft + ((size_t)slot * BN + mb * 256 + wv * 32 + mn) * KMIX + kh;
  const __bf16* ap1 = ap0 + (size_t)16 * KMIX;
  v8f acc[2][4]; zero_acc(acc);

  const int crow = threadIdx.x >> 2, cseg = threadIdx.x & 3;  // copy: 64 rows x 4 segs
  for (int half = 0; half < 2; ++half) {
    // async copy one K-half of the B panel: 64 rows x 256 elems (512B/row) -> LDS
    {
      const __bf16* grow = bsrc + (size_t)crow * KMIX + half * 256;
      const uint32_t lrow = lbase + (uint32_t)(crow * (BPAD * 2));
#pragma unroll
      for (int i = 0; i < 8; ++i) {
        const uint32_t boff = (uint32_t)(cseg * 128 + i * 16);
        const __bf16* g = (const __bf16*)((const char*)grow + boff);
        asm volatile("global_load_async_to_lds_b128 %0, %1, off"
                     :: "v"(lrow + boff), "v"(g) : "memory");
      }
      asm volatile("s_wait_asynccnt 0x0" ::: "memory");
    }
    __syncthreads();
#pragma unroll 2
    for (int kk = 0; kk < 256; kk += 32) {
      v16bf a0 = *(const v16bf*)(ap0 + half * 256 + kk);
      v16bf a1 = *(const v16bf*)(ap1 + half * 256 + kk);
#pragma unroll
      for (int ni = 0; ni < 4; ++ni) {
        v16bf b = *(const v16bf*)(&Bpan[(ni * 16 + mn) * BPAD + kh + kk]);   // ds_load_b128
        acc[0][ni] = wmma_bf16(a0, b, acc[0][ni]);
        acc[1][ni] = wmma_bf16(a1, b, acc[1][ni]);
      }
    }
    __syncthreads();
  }

  const int rb = (lane >> 4) << 3;
#pragma unroll
  for (int ni = 0; ni < 4; ++ni) {
    const int col = nb * 64 + ni * 16 + (lane & 15);
    const int h = col >> 6, opart = (col >> 5) & 1, o = col & 31;
    const int kidx = slot * 2 + opart;
#pragma unroll
    for (int mi = 0; mi < 2; ++mi)
#pragma unroll
      for (int r = 0; r < 8; ++r) {
        const int bn = mb * 256 + wv * 32 + mi * 16 + rb + r;
        Ybf[(((size_t)bn * H_ + h) * DK_ + o) * 256 + kidx] = (__bf16)acc[mi][ni][r];
      }
  }
}

// ---------------- stage C: inverse DFT GEMM [131072x256]@[256x256] ----------------
// vmode==0: scatter as A-layout [bh][n][o*256+s]   (Q, K -> D1 operands)
// vmode==1: scatter as B^T layout [bh][o*256+s][n] (V -> D2 B operand)
__global__ void stageC_ifft(const __bf16* __restrict__ Ybf,
                            const __bf16* __restrict__ BasT,
                            __bf16* __restrict__ sp, int vmode) {
  const int w = blockIdx.x * (blockDim.x >> 5) + (threadIdx.x >> 5);
  const int lane = threadIdx.x & 31;
  const int nm = w & 3, mm = w >> 2;
  v8f acc[2][4]; zero_acc(acc);
  wmma_block2x4(Ybf + (size_t)mm * 32 * 256, 256, BasT + (size_t)nm * 64 * 256, 256, 256, lane, acc);
  const int rb = (lane >> 4) << 3;
#pragma unroll
  for (int ni = 0; ni < 4; ++ni) {
    const int s = nm * 64 + ni * 16 + (lane & 15);
#pragma unroll
    for (int mi = 0; mi < 2; ++mi)
#pragma unroll
      for (int r = 0; r < 8; ++r) {
        const int row = mm * 32 + mi * 16 + rb + r;       // ((bn*8+h)*32+o)
        const int o = row & 31, h = (row >> 5) & 7, bn = row >> 8;
        const int b = bn >> 8, n = bn & 255;
        const __bf16 val = (__bf16)acc[mi][ni][r];
        if (vmode == 0)
          sp[(((size_t)(b * H_ + h) * N_ + n) * FEAT) + o * S_ + s] = val;
        else
          sp[(((size_t)(b * H_ + h) * FEAT) + o * S_ + s) * N_ + n] = val;
      }
  }
}

// ---------------- stage D1: scores = Q K^T / (P*P), K = 8192 ----------------
__global__ void stageD1_scores(const __bf16* __restrict__ Qsp,
                               const __bf16* __restrict__ Ksp,
                               float* __restrict__ scores) {
  const int w = blockIdx.x * (blockDim.x >> 5) + (threadIdx.x >> 5);
  const int lane = threadIdx.x & 31;
  const int nm = w & 3, mm = (w >> 2) & 7, bh = w >> 5;   // 16 x 8 x 4
  v8f acc[2][4]; zero_acc(acc);
  wmma_block2x4(Qsp + ((size_t)bh * N_ + mm * 32) * FEAT, FEAT,
                Ksp + ((size_t)bh * N_ + nm * 64) * FEAT, FEAT, FEAT, lane, acc);
  const int rb = (lane >> 4) << 3;
#pragma unroll
  for (int ni = 0; ni < 4; ++ni) {
    const int col = nm * 64 + ni * 16 + (lane & 15);
#pragma unroll
    for (int mi = 0; mi < 2; ++mi)
#pragma unroll
      for (int r = 0; r < 8; ++r) {
        const int row = mm * 32 + mi * 16 + rb + r;
        scores[((size_t)bh * N_ + row) * N_ + col] = acc[mi][ni][r] * INV_SCALE;
      }
  }
}

// ---------------- softmax over last dim (256) ----------------
__global__ void softmax_rows(const float* __restrict__ scores, __bf16* __restrict__ attn) {
  __shared__ float red[256];
  const int row = blockIdx.x, t = threadIdx.x;
  const float v = scores[(size_t)row * 256 + t];
  red[t] = v; __syncthreads();
  for (int s = 128; s > 0; s >>= 1) { if (t < s) red[t] = fmaxf(red[t], red[t + s]); __syncthreads(); }
  const float m = red[0]; __syncthreads();
  const float e = __expf(v - m);
  red[t] = e; __syncthreads();
  for (int s = 128; s > 0; s >>= 1) { if (t < s) red[t] += red[t + s]; __syncthreads(); }
  attn[(size_t)row * 256 + t] = (__bf16)(e / red[0]);
}

// ---------------- stage D2: out = attn @ V (V stored B-transposed), scatter to Feat --
// per (b,h): M=256(n), N=8192(c=o*256+s), K=256(n'); both operands contiguous 32B frags.
__global__ void stageD2_av(const __bf16* __restrict__ attn,
                           const __bf16* __restrict__ VspT,
                           __bf16* __restrict__ Feat) {
  const int w = blockIdx.x * (blockDim.x >> 5) + (threadIdx.x >> 5);
  const int lane = threadIdx.x & 31;
  const int nm = w & 127, mm = (w >> 7) & 7, bh = w >> 10;  // 16 x 8 x 128
  v8f acc[2][4]; zero_acc(acc);
  wmma_block2x4(attn + ((size_t)bh * N_ + mm * 32) * 256, 256,
                VspT + ((size_t)bh * FEAT + nm * 64) * 256, 256, 256, lane, acc);
  const int b_ = bh >> 3, h = bh & 7;
  const int rb = (lane >> 4) << 3;
#pragma unroll
  for (int ni = 0; ni < 4; ++ni) {
    const int c = nm * 64 + ni * 16 + (lane & 15);
    const int o = c >> 8, s = c & 255;
#pragma unroll
    for (int mi = 0; mi < 2; ++mi)
#pragma unroll
      for (int r = 0; r < 8; ++r) {
        const int n = mm * 32 + mi * 16 + rb + r;
        Feat[(((size_t)(b_ * N_ + n) * S_ + s) << 8) + o * H_ + h] = (__bf16)acc[mi][ni][r];
      }
  }
}

// ---------------- stage E: output projection + bias ----------------
__global__ void stageE_proj(const __bf16* __restrict__ Feat,
                            const __bf16* __restrict__ WoutT,
                            const float* __restrict__ b_out,
                            float* __restrict__ out) {
  const int w = blockIdx.x * (blockDim.x >> 5) + (threadIdx.x >> 5);
  const int lane = threadIdx.x & 31;
  const int nm = w & 3, mm = w >> 2;
  v8f acc[2][4]; zero_acc(acc);
  wmma_block2x4(Feat + (size_t)mm * 32 * 256, 256, WoutT + (size_t)nm * 64 * 256, 256, 256, lane, acc);
  const int rb = (lane >> 4) << 3;
#pragma unroll
  for (int ni = 0; ni < 4; ++ni) {
    const int col = nm * 64 + ni * 16 + (lane & 15);
    const float bias = b_out[col];
#pragma unroll
    for (int mi = 0; mi < 2; ++mi)
#pragma unroll
      for (int r = 0; r < 8; ++r)
        out[((size_t)(mm * 32 + mi * 16 + rb + r) << 8) + col] = acc[mi][ni][r] + bias;
  }
}

// ---------------- launch ----------------
extern "C" void kernel_launch(void* const* d_in, const int* in_sizes, int n_in,
                              void* d_out, int out_size, void* d_ws, size_t ws_size,
                              hipStream_t stream) {
  (void)in_sizes; (void)n_in; (void)out_size; (void)ws_size;
  const float* x = (const float*)d_in[0];
  const float* wspec[3][2] = {
    {(const float*)d_in[1], (const float*)d_in[2]},   // q
    {(const float*)d_in[3], (const float*)d_in[4]},   // k
    {(const float*)d_in[5], (const float*)d_in[6]},   // v
  };
  const float* w_out = (const float*)d_in[7];
  const float* b_out = (const float*)d_in[8];
  char* ws = (char*)d_ws;

  __bf16* xT    = (__bf16*)(ws + OFF_XT);
  __bf16* Xft   = (__bf16*)(ws + OFF_XFT);
  __bf16* WbigT = (__bf16*)(ws + OFF_WBIG);
  __bf16* Ybf   = (__bf16*)(ws + OFF_YBF);
  __bf16* sp[3] = {(__bf16*)(ws + OFF_QSP), (__bf16*)(ws + OFF_KSP), (__bf16*)(ws + OFF_VSPT)};
  float*  scor  = (float*)(ws + OFF_SCORE);
  __bf16* attn  = (__bf16*)(ws + OFF_ATTN);
  __bf16* FwdT  = (__bf16*)(ws + OFF_FWDT);
  __bf16* BasT  = (__bf16*)(ws + OFF_BAST);
  __bf16* WoutT = (__bf16*)(ws + OFF_WOUT);
  __bf16* Feat  = (__bf16*)(ws + OFF_FEAT);

  init_basis<<<256, 256, 0, stream>>>(w_out, FwdT, BasT, WoutT);
  transpose_x<<<BN, 256, 0, stream>>>(x, xT);
  stageA_fft<<<2048, 256, 0, stream>>>(xT, FwdT, Xft);

  for (int t = 0; t < 3; ++t) {
    pack_w<<<131072, 256, 0, stream>>>(wspec[t][0], wspec[t][1], WbigT);
    stageB_mix<<<2048, 256, 0, stream>>>(Xft, WbigT, Ybf);
    stageC_ifft<<<2048, 256, 0, stream>>>(Ybf, BasT, sp[t], (t == 2) ? 1 : 0);
  }

  stageD1_scores<<<64, 256, 0, stream>>>(sp[0], sp[1], scor);
  softmax_rows<<<16 * 256, 256, 0, stream>>>(scor, attn);
  stageD2_av<<<2048, 256, 0, stream>>>(attn, sp[2], Feat);
  stageE_proj<<<2048, 256, 0, stream>>>(Feat, WoutT, b_out, (float*)d_out);
}
